// LSTMPredictor_51213190037957
// MI455X (gfx1250) — matrix operand order repaired and checked by reading, not compiled, not run on hardware
//
#include <hip/hip_runtime.h>
#include <stdint.h>

// ---------------- problem dims ----------------
#define S_LEN 16384
#define D_IN  256
#define H_DIM 1024
#define T_OUT 64
#define G4    (4 * H_DIM)  // 4096

typedef __attribute__((ext_vector_type(16))) _Float16 v16h;
typedef __attribute__((ext_vector_type(8)))  _Float16 v8h;
typedef __attribute__((ext_vector_type(8)))  float    v8f;
typedef int v4i __attribute__((vector_size(16)));

// generic -> concrete address-space typed-pointer casts (HIP/clang addrspacecast)
#define AS1P(T, p) ((__attribute__((address_space(1))) T*)(p))
#define AS3P(T, p) ((__attribute__((address_space(3))) T*)(p))

#if __has_builtin(__builtin_amdgcn_global_load_async_to_lds_b128) && \
    __has_builtin(__builtin_amdgcn_global_load_async_to_lds_b32)
#define HAVE_ASYNC_LDS 1
#else
#define HAVE_ASYNC_LDS 0
#endif

__device__ inline void wait_asynccnt0() {
#if __has_builtin(__builtin_amdgcn_s_wait_asynccnt)
    __builtin_amdgcn_s_wait_asynccnt(0);
#else
    asm volatile("s_wait_asynccnt 0x0" ::: "memory");
#endif
}

// ---------------- WMMA fragment loaders (wave32, contiguous-K order used for both A and B) ----------------
// A (16x32 f16, row-major source, leading dim lda):
//   lane L<16 : row L,  K = k0+{0..15}
//   lane L>=16: row L-16, K = k0+{16..31}
__device__ inline v16h load_a_frag(const _Float16* __restrict__ A, int lda,
                                   int row0, int k0, int lane) {
    int r  = row0 + (lane & 15);
    int ks = k0 + ((lane >> 4) << 4);  // +0 or +16
    const v8h* p = (const v8h*)(A + (size_t)r * lda + ks);
    v8h lo = p[0];  // K = ks .. ks+7
    v8h hi = p[1];  // K = ks+8 .. ks+15
    return __builtin_shufflevector(lo, hi, 0, 1, 2, 3, 4, 5, 6, 7,
                                   8, 9, 10, 11, 12, 13, 14, 15);
}

// B (32x16 f16) where B = W^T and W is row-major (N x K, leading dim ldw):
//   lane n<16 : column n  = W row (n0+n), K = k0..k0+15   (32 contiguous bytes)
//   lane n>=16: column n-16,              K = k0+16..k0+31
__device__ inline v16h load_b_frag(const _Float16* __restrict__ W, int ldw,
                                   int n0, int k0, int lane) {
    int n  = n0 + (lane & 15);
    int kk = k0 + ((lane >> 4) << 4);  // +0 or +16
    return *(const v16h*)(W + (size_t)n * ldw + kk);
}

// ---------------- conversion / init kernels ----------------
__global__ void cvt_f32_f16(const float* __restrict__ src,
                            _Float16* __restrict__ dst, int n) {
    int i = blockIdx.x * blockDim.x + threadIdx.x;
    int stride = gridDim.x * blockDim.x;
    for (; i < n; i += stride) dst[i] = (_Float16)src[i];
}

__global__ void init_misc(const float* __restrict__ b_ih,
                          const float* __restrict__ b_hh,
                          float* __restrict__ bias,
                          _Float16* __restrict__ hs0,
                          unsigned* __restrict__ ctr) {
    int i = blockIdx.x * blockDim.x + threadIdx.x;
    if (i < G4)   bias[i] = b_ih[i] + b_hh[i];
    if (i < H_DIM) hs0[i] = (_Float16)0.f;
    if (i == 0)   *ctr = 0u;
}

// ---------------- phase 1: x_proj = X * W_ih^T + bias ----------------
// grid: (4096/64, 16384/128) ; block 256 (8 waves). Each wave: 16x64 tile.
__global__ void __launch_bounds__(256)
xproj_gemm(const _Float16* __restrict__ X,
           const _Float16* __restrict__ Wih,
           const float* __restrict__ bias,
           float* __restrict__ XP) {
    int lane = threadIdx.x & 31;
    int wave = threadIdx.x >> 5;
    int n0   = blockIdx.x * 64;
    int row0 = blockIdx.y * 128 + wave * 16;

    v8f acc[4] = {v8f{}, v8f{}, v8f{}, v8f{}};
#pragma unroll
    for (int k0 = 0; k0 < D_IN; k0 += 32) {
        v16h a = load_a_frag(X, D_IN, row0, k0, lane);
#pragma unroll
        for (int tn = 0; tn < 4; ++tn) {
            v16h b = load_b_frag(Wih, D_IN, n0 + tn * 16, k0, lane);
            acc[tn] = __builtin_amdgcn_wmma_f32_16x16x32_f16(
                false, a, false, b, (short)0, acc[tn], false, false);
        }
    }
    int nc = lane & 15;
    int rb = row0 + ((lane >> 4) << 3);
#pragma unroll
    for (int tn = 0; tn < 4; ++tn) {
        int col = n0 + tn * 16 + nc;
        float bv = bias[col];
#pragma unroll
        for (int g = 0; g < 8; ++g) {
            XP[(size_t)(rb + g) * G4 + col] = acc[tn][g] + bv;
        }
    }
}

// ---------------- phase 2: persistent LSTM recurrence ----------------
// 64 workgroups x 512 threads (16 waves). WG b owns h-indices j in [16b,16b+16).
// LDS: f16 W_hh slice [4 gates][16 j][1024 k] = 128KB, + h broadcast (2KB).
#define NWG  64
#define RBLK 512

__global__ void __launch_bounds__(RBLK)
lstm_recur(const float* __restrict__ XP,
           const _Float16* __restrict__ Whh,
           _Float16* __restrict__ hs,     // (S+1) x H, row 0 pre-zeroed
           unsigned* __restrict__ ctr) {
    extern __shared__ _Float16 smem[];
    _Float16* sW = smem;               // [64][1024]
    _Float16* sh = smem + 64 * 1024;   // [1024]

    int tid  = threadIdx.x;
    int lane = tid & 31;
    int wave = tid >> 5;
    int j0   = blockIdx.x * 16;
    int j    = j0 + wave;

    // Preload this WG's 64 W_hh rows into LDS (rows: gate*1024 + j0 + jj).
    // CDNA5 path: GLOBAL_LOAD_ASYNC_TO_LDS_B128 (no VGPR staging, ASYNCcnt).
    for (int idx = tid; idx < (64 * 1024) / 8; idx += RBLK) {
        int rl = idx >> 7;            // local row 0..63
        int kc = (idx & 127) << 3;    // k chunk start
        int grow = (rl >> 4) * H_DIM + j0 + (rl & 15);
#if HAVE_ASYNC_LDS
        __builtin_amdgcn_global_load_async_to_lds_b128(
            AS1P(v4i, Whh + (size_t)grow * H_DIM + kc),
            AS3P(v4i, sW + rl * 1024 + kc), 0, 0);
#else
        *(v8h*)(sW + rl * 1024 + kc) =
            *(const v8h*)(Whh + (size_t)grow * H_DIM + kc);
#endif
    }
#if HAVE_ASYNC_LDS
    wait_asynccnt0();
#endif
    __syncthreads();

    float cst = 0.f;  // cell state c[j], lives in lane 0 of each wave

    for (int t = 0; t < S_LEN; ++t) {
        // broadcast h_t into LDS (512 threads x 1 dword = 1024 f16)
#if HAVE_ASYNC_LDS
        __builtin_amdgcn_global_load_async_to_lds_b32(
            AS1P(int, (const unsigned*)(hs + (size_t)t * H_DIM) + tid),
            AS3P(int, (unsigned*)sh + tid), 0, 0);
        wait_asynccnt0();
#else
        ((unsigned*)sh)[tid] =
            ((const unsigned*)(hs + (size_t)t * H_DIM))[tid];
#endif
        __syncthreads();

        float a0 = 0.f, a1 = 0.f, a2 = 0.f, a3 = 0.f;
#pragma unroll
        for (int it = 0; it < 4; ++it) {
            int k = (lane << 3) + (it << 8);
            v8h hv = *(const v8h*)(sh + k);
            v8h w0 = *(const v8h*)(sW + (0 * 16 + wave) * 1024 + k);
            v8h w1 = *(const v8h*)(sW + (1 * 16 + wave) * 1024 + k);
            v8h w2 = *(const v8h*)(sW + (2 * 16 + wave) * 1024 + k);
            v8h w3 = *(const v8h*)(sW + (3 * 16 + wave) * 1024 + k);
#pragma unroll
            for (int e = 0; e < 8; ++e) {
                float hf = (float)hv[e];
                a0 += hf * (float)w0[e];
                a1 += hf * (float)w1[e];
                a2 += hf * (float)w2[e];
                a3 += hf * (float)w3[e];
            }
        }
        // wave32 all-reduce
#pragma unroll
        for (int m = 16; m >= 1; m >>= 1) {
            a0 += __shfl_xor(a0, m, 32);
            a1 += __shfl_xor(a1, m, 32);
            a2 += __shfl_xor(a2, m, 32);
            a3 += __shfl_xor(a3, m, 32);
        }

        if (lane == 0) {
            const float* xp = XP + (size_t)t * G4;
            float gi = xp[j] + a0;
            float gf = xp[H_DIM + j] + a1;
            float gg = xp[2 * H_DIM + j] + a2;
            float go = xp[3 * H_DIM + j] + a3;
            gi = 1.f / (1.f + __expf(-gi));
            gf = 1.f / (1.f + __expf(-gf));
            gg = tanhf(gg);
            go = 1.f / (1.f + __expf(-go));
            cst = gf * cst + gi * gg;
            float hnew = go * tanhf(cst);
            hs[(size_t)(t + 1) * H_DIM + j] = (_Float16)hnew;
            __threadfence();  // release this wave's h write
            if (t + 1 < S_LEN) {  // warm L2/GL1 for next step's gate biases
                const float* xpn = XP + (size_t)(t + 1) * G4;
                __builtin_prefetch(xpn + j, 0, 1);
                __builtin_prefetch(xpn + H_DIM + j, 0, 1);
                __builtin_prefetch(xpn + 2 * H_DIM + j, 0, 1);
                __builtin_prefetch(xpn + 3 * H_DIM + j, 0, 1);
            }
        }
        __syncthreads();

        // grid-wide step barrier (monotonic counter, no reset race)
        if (tid == 0) {
            atomicAdd(ctr, 1u);
            unsigned target = (unsigned)(t + 1) * (unsigned)NWG;
            while (__hip_atomic_load(ctr, __ATOMIC_ACQUIRE,
                                     __HIP_MEMORY_SCOPE_AGENT) < target) {
                __builtin_amdgcn_s_sleep(2);
            }
        }
        __syncthreads();
    }
}

// ---------------- phase 3: logits = HS * W_out^T + b ; row softmax ----------------
// grid: 16384/128 blocks; block 256 (8 waves); wave => 16x64 tile (full T).
__global__ void __launch_bounds__(256)
out_softmax(const _Float16* __restrict__ HS,   // hs + H (rows 1..S)
            const _Float16* __restrict__ Wout,
            const float* __restrict__ bout,
            float* __restrict__ OUT) {
    int lane = threadIdx.x & 31;
    int wave = threadIdx.x >> 5;
    int row0 = (blockIdx.x * 8 + wave) * 16;

    v8f acc[4] = {v8f{}, v8f{}, v8f{}, v8f{}};
    for (int k0 = 0; k0 < H_DIM; k0 += 32) {
        v16h a = load_a_frag(HS, H_DIM, row0, k0, lane);
#pragma unroll
        for (int tn = 0; tn < 4; ++tn) {
            v16h b = load_b_frag(Wout, H_DIM, tn * 16, k0, lane);
            acc[tn] = __builtin_amdgcn_wmma_f32_16x16x32_f16(
                false, a, false, b, (short)0, acc[tn], false, false);
        }
    }

    int nc = lane & 15;
    int rb = row0 + ((lane >> 4) << 3);
    float bv[4];
#pragma unroll
    for (int tn = 0; tn < 4; ++tn) bv[tn] = bout[tn * 16 + nc];

#pragma unroll
    for (int g = 0; g < 8; ++g) {
        float v[4];
        float m = -3.0e38f;
#pragma unroll
        for (int tn = 0; tn < 4; ++tn) {
            v[tn] = acc[tn][g] + bv[tn];
            m = fmaxf(m, v[tn]);
        }
        // reduce across the 16 lanes holding this row's 64 columns
#pragma unroll
        for (int x = 8; x >= 1; x >>= 1) m = fmaxf(m, __shfl_xor(m, x, 32));
        float s = 0.f;
#pragma unroll
        for (int tn = 0; tn < 4; ++tn) {
            v[tn] = __expf(v[tn] - m);
            s += v[tn];
        }
#pragma unroll
        for (int x = 8; x >= 1; x >>= 1) s += __shfl_xor(s, x, 32);
        float inv = 1.f / s;
        int row = rb + g;
#pragma unroll
        for (int tn = 0; tn < 4; ++tn) {
            OUT[(size_t)row * T_OUT + tn * 16 + nc] = v[tn] * inv;
        }
    }
}

// ---------------- workspace layout (bytes, 256-aligned) ----------------
#define XH_OFF    ((size_t)0)               //  8,388,608 : x in f16
#define WIH_OFF   ((size_t)8388608)         //  2,097,152 : W_ih f16
#define WHH_OFF   ((size_t)10485760)        //  8,388,608 : W_hh f16
#define WOUT_OFF  ((size_t)18874368)        //    131,072 : W_out f16
#define BIAS_OFF  ((size_t)19005440)        //     16,384 : b_ih + b_hh
#define CTR_OFF   ((size_t)19021824)        //        256 : barrier counter
#define HS_OFF    ((size_t)19022080)        // 33,556,480 : (S+1) x H f16 history
#define XP_OFF    ((size_t)52578560)        // 268,435,456: x_proj fp32

extern "C" void kernel_launch(void* const* d_in, const int* in_sizes, int n_in,
                              void* d_out, int out_size, void* d_ws, size_t ws_size,
                              hipStream_t stream) {
    (void)in_sizes; (void)n_in; (void)out_size; (void)ws_size;

    const float* x     = (const float*)d_in[0];
    const float* W_ih  = (const float*)d_in[1];
    const float* W_hh  = (const float*)d_in[2];
    const float* b_ih  = (const float*)d_in[3];
    const float* b_hh  = (const float*)d_in[4];
    const float* W_out = (const float*)d_in[5];
    const float* b_out = (const float*)d_in[6];
    float* out = (float*)d_out;

    char* ws = (char*)d_ws;
    _Float16* xh    = (_Float16*)(ws + XH_OFF);
    _Float16* wih_h = (_Float16*)(ws + WIH_OFF);
    _Float16* whh_h = (_Float16*)(ws + WHH_OFF);
    _Float16* wout_h= (_Float16*)(ws + WOUT_OFF);
    float*    bias  = (float*)   (ws + BIAS_OFF);
    unsigned* ctr   = (unsigned*)(ws + CTR_OFF);
    _Float16* hs    = (_Float16*)(ws + HS_OFF);
    float*    xp    = (float*)   (ws + XP_OFF);

    // fp32 -> f16 conversions
    cvt_f32_f16<<<2048, 256, 0, stream>>>(x,     xh,     S_LEN * D_IN);
    cvt_f32_f16<<<1024, 256, 0, stream>>>(W_ih,  wih_h,  G4 * D_IN);
    cvt_f32_f16<<<2048, 256, 0, stream>>>(W_hh,  whh_h,  G4 * H_DIM);
    cvt_f32_f16<<<256,  256, 0, stream>>>(W_out, wout_h, T_OUT * H_DIM);
    init_misc  <<<16,   256, 0, stream>>>(b_ih, b_hh, bias, hs, ctr);

    // phase 1: batched input projection (WMMA)
    xproj_gemm<<<dim3(G4 / 64, S_LEN / 128), 256, 0, stream>>>(xh, wih_h, bias, xp);

    // phase 2: persistent recurrence, W_hh resident in LDS (async-to-LDS path)
    size_t lds_bytes = (64 * 1024 + 1024) * sizeof(_Float16);  // 133,120 B
    lstm_recur<<<NWG, RBLK, lds_bytes, stream>>>(xp, whh_h, hs, ctr);

    // phase 3: output projection + fused softmax (WMMA)
    out_softmax<<<S_LEN / 128, 256, 0, stream>>>(hs + H_DIM, wout_h, b_out, out);
}
